// MambaReverseRefinement_15891378995798
// MI455X (gfx1250) — matrix operand order repaired and checked by reading, not compiled, not run on hardware
//
#include <hip/hip_runtime.h>
#include <hip/hip_bf16.h>

typedef __attribute__((ext_vector_type(16))) _Float16 v16h;
typedef __attribute__((ext_vector_type(8)))  _Float16 v8h;
typedef __attribute__((ext_vector_type(8)))  float    v8f;

#define DIMC    96
#define HH      48
#define WWD     48
#define HWSZ    2304
#define BATCH   4
#define NTOK    9216      // BATCH * HWSZ
#define DINNER  192
#define NDIR    4
#define DSTATE  16
#define DTRANK  6
#define CDBL    38        // DTRANK + 2*DSTATE
#define LSEQ    2304
#define BN_EPS  1e-3f
#define LN_EPS  1e-5f

// ---------------------------------------------------------------- helpers
__device__ __forceinline__ float geluf(float x) {
    return 0.5f * x * (1.f + erff(x * 0.7071067811865476f));
}
__device__ __forceinline__ float siluf(float x) {
    return x / (1.f + __expf(-x));
}
__device__ __forceinline__ float softplusf(float x) {
    return x > 20.f ? x : log1pf(__expf(x));
}
// token (h*W+w) for sequence position l of direction k
__device__ __forceinline__ int tok_of(int k, int l) {
    int ll = (k & 2) ? (LSEQ - 1 - l) : l;
    if (k & 1) { int w = ll / HH; int h = ll % HH; return h * WWD + w; }
    return ll;
}

// ------------------------------------------------------- WMMA tile engine
// C[16x16] += A[16x32] * B[32x16] over K, A row-major [M][K], W row-major [N][K]
template <int KK>
__device__ __forceinline__ v8f gemm_tile(const _Float16* __restrict__ A,
                                         const _Float16* __restrict__ W,
                                         int mt, int nt, int lane) {
    v8f acc;
#pragma unroll
    for (int i = 0; i < 8; ++i) acc[i] = 0.f;
    const int hs = lane >> 4;                    // lane half select
    const _Float16* arow = A + (size_t)(mt * 16 + (lane & 15)) * KK;
    const _Float16* wrow = W + (size_t)(nt * 16 + (lane & 15)) * KK;
#pragma unroll
    for (int kc = 0; kc < KK; kc += 32) {
        const int ka = kc + (hs << 3);           // A: +0 / +8
        const int kb = kc + (hs << 3);           // B: +0 / +8
        v8h a0 = *(const v8h*)(arow + ka);
        v8h a1 = *(const v8h*)(arow + ka + 16);
        v8h b0 = *(const v8h*)(wrow + kb);
        v8h b1 = *(const v8h*)(wrow + kb + 16);
        v16h av, bv;
#pragma unroll
        for (int i = 0; i < 8; ++i) {
            av[i] = a0[i]; av[i + 8] = a1[i];
            bv[i] = b0[i]; bv[i + 8] = b1[i];
        }
        acc = __builtin_amdgcn_wmma_f32_16x16x32_f16(
            false, av, false, bv, (short)0, acc, false, false);
    }
    return acc;
}

// ------------------------------------------------------------- kernels
__global__ void cvt_f16_kernel(const float* __restrict__ src,
                               _Float16* __restrict__ dst, int n) {
    int i = blockIdx.x * blockDim.x + threadIdx.x;
    if (i < n) dst[i] = (_Float16)src[i];
}

// reverse-attention mask + transpose (B,C,H,W)->(tok,C) + f16
__global__ void mask_kernel(const float* __restrict__ feature,
                            const float* __restrict__ pred,
                            const float* __restrict__ bg, const float* __restrict__ bb,
                            const float* __restrict__ bm, const float* __restrict__ bv,
                            _Float16* __restrict__ fm16) {
    int i = blockIdx.x * blockDim.x + threadIdx.x;
    if (i >= NTOK * DIMC) return;
    int t = i / DIMC, c = i % DIMC;
    int b = t / HWSZ, hw = t % HWSZ;
    float sc = bg[0] * rsqrtf(bv[0] + BN_EPS);
    float pn = (pred[b * HWSZ + hw] - bm[0]) * sc + bb[0];
    float rm = 1.f - 1.f / (1.f + __expf(-pn));
    float f = feature[((size_t)b * DIMC + c) * HWSZ + hw];
    fm16[i] = (_Float16)(f * rm);
}

// pre GEMM (96x96) + BN + exact GELU -> f32 token-major
__global__ void gemm_pre_kernel(const _Float16* __restrict__ A,
                                const _Float16* __restrict__ W,
                                const float* __restrict__ g, const float* __restrict__ b,
                                const float* __restrict__ m, const float* __restrict__ v,
                                float* __restrict__ out) {
    const int lane = threadIdx.x & 31, wid = threadIdx.x >> 5;
    const int tilesN = DIMC / 16, tot = (NTOK / 16) * tilesN;
    int tile = blockIdx.x * 8 + wid;
    if (tile >= tot) return;
    int mt = tile / tilesN, nt = tile % tilesN;
    v8f acc = gemm_tile<DIMC>(A, W, mt, nt, lane);
    int col = nt * 16 + (lane & 15);
    float sc = g[col] * rsqrtf(v[col] + BN_EPS);
#pragma unroll
    for (int r = 0; r < 8; ++r) {
        int row = mt * 16 + r + ((lane >> 4) << 3);
        float x = (acc[r] - m[col]) * sc + b[col];
        out[(size_t)row * DIMC + col] = geluf(x);
    }
}

// layernorm over 96 channels -> f16
__global__ void ln96_kernel(const float* __restrict__ in,
                            const float* __restrict__ g, const float* __restrict__ b,
                            _Float16* __restrict__ out) {
    int t = blockIdx.x * blockDim.x + threadIdx.x;
    if (t >= NTOK) return;
    const float* row = in + (size_t)t * DIMC;
    float s = 0.f, s2 = 0.f;
    for (int c = 0; c < DIMC; ++c) { float x = row[c]; s += x; s2 += x * x; }
    float mu = s / DIMC;
    float inv = rsqrtf(s2 / DIMC - mu * mu + LN_EPS);
    for (int c = 0; c < DIMC; ++c)
        out[(size_t)t * DIMC + c] = (_Float16)(((row[c] - mu) * inv) * g[c] + b[c]);
}

// in_proj GEMM (384x96) -> raw f32 xz
__global__ void gemm_in_kernel(const _Float16* __restrict__ A,
                               const _Float16* __restrict__ W,
                               float* __restrict__ out) {
    const int lane = threadIdx.x & 31, wid = threadIdx.x >> 5;
    const int tilesN = (2 * DINNER) / 16, tot = (NTOK / 16) * tilesN;
    int tile = blockIdx.x * 8 + wid;
    if (tile >= tot) return;
    int mt = tile / tilesN, nt = tile % tilesN;
    v8f acc = gemm_tile<DIMC>(A, W, mt, nt, lane);
    int col = nt * 16 + (lane & 15);
#pragma unroll
    for (int r = 0; r < 8; ++r) {
        int row = mt * 16 + r + ((lane >> 4) << 3);
        out[(size_t)row * (2 * DINNER) + col] = acc[r];
    }
}

// depthwise 3x3 conv + bias + SiLU; reads xz[:, :192] token-major, writes channel-major
__global__ void conv_kernel(const float* __restrict__ xz,
                            const float* __restrict__ w,
                            const float* __restrict__ bias,
                            float* __restrict__ out) {
    int i = blockIdx.x * blockDim.x + threadIdx.x;
    if (i >= BATCH * DINNER * HWSZ) return;
    int hw = i % HWSZ, d = (i / HWSZ) % DINNER, b = i / (HWSZ * DINNER);
    int h = hw / WWD, x = hw % WWD;
    float acc = bias[d];
#pragma unroll
    for (int ky = 0; ky < 3; ++ky) {
#pragma unroll
        for (int kx = 0; kx < 3; ++kx) {
            int yy = h + ky - 1, xx = x + kx - 1;
            if (yy >= 0 && yy < HH && xx >= 0 && xx < WWD) {
                acc += xz[((size_t)(b * HWSZ + yy * WWD + xx)) * (2 * DINNER) + d]
                       * w[d * 9 + ky * 3 + kx];
            }
        }
    }
    out[((size_t)(b * DINNER + d)) * HWSZ + hw] = siluf(acc);
}

// x_proj: one thread per (b,k,l), all 38 outputs accumulated in registers
__global__ void xproj_kernel(const float* __restrict__ xc,
                             const float* __restrict__ w,
                             float* __restrict__ xdbl) {
    int i = blockIdx.x * blockDim.x + threadIdx.x;
    if (i >= BATCH * NDIR * LSEQ) return;
    int l = i % LSEQ, k = (i / LSEQ) % NDIR, b = i / (LSEQ * NDIR);
    int tok = tok_of(k, l);
    float acc[CDBL];
#pragma unroll
    for (int c = 0; c < CDBL; ++c) acc[c] = 0.f;
    const float* wk = w + (size_t)k * CDBL * DINNER;
    for (int d = 0; d < DINNER; ++d) {
        float xv = xc[((size_t)(b * DINNER + d)) * HWSZ + tok];
#pragma unroll
        for (int c = 0; c < CDBL; ++c) acc[c] += xv * wk[c * DINNER + d];
    }
    float* o = xdbl + ((size_t)(b * NDIR + k) * CDBL) * LSEQ + l;
#pragma unroll
    for (int c = 0; c < CDBL; ++c) o[(size_t)c * LSEQ] = acc[c];
}

// selective scan: one workgroup per (b,k); 192 threads = one channel each.
#define SCH 32
__global__ void __launch_bounds__(DINNER)
scan_kernel(const float* __restrict__ xc,     // (b,d,hw)
            const float* __restrict__ xdbl,   // (b,k,38,l)
            const float* __restrict__ dt_w,   // (4,192,6)
            const float* __restrict__ dt_b,   // (4,192)
            const float* __restrict__ A_logs, // (768,16)
            const float* __restrict__ Ds,     // (768,)
            float* __restrict__ ys) {         // (b,k,d,l)
    __shared__ float s_bc[CDBL * SCH];
    __shared__ float s_u[DINNER * SCH];
    const int d = threadIdx.x;
    const int b = blockIdx.x / NDIR, k = blockIdx.x % NDIR;
    const int ch = k * DINNER + d;
    float Av[DSTATE], hst[DSTATE], dtw[DTRANK];
#pragma unroll
    for (int n = 0; n < DSTATE; ++n) {
        Av[n] = -__expf(A_logs[(size_t)ch * DSTATE + n]);
        hst[n] = 0.f;
    }
#pragma unroll
    for (int r = 0; r < DTRANK; ++r) dtw[r] = dt_w[(size_t)ch * DTRANK + r];
    const float Dd = Ds[ch], dtb = dt_b[ch];
    const float* xdk = xdbl + (size_t)(b * NDIR + k) * CDBL * LSEQ;
    float* yo = ys + ((size_t)(b * NDIR + k) * DINNER + d) * LSEQ;

    for (int c0 = 0; c0 < LSEQ; c0 += SCH) {
        for (int idx = d; idx < CDBL * SCH; idx += DINNER) {
            int c = idx / SCH, lc = idx % SCH;
            s_bc[idx] = xdk[(size_t)c * LSEQ + c0 + lc];
        }
        for (int idx = d; idx < DINNER * SCH; idx += DINNER) {
            int dd = idx / SCH, lc = idx % SCH;
            s_u[idx] = xc[((size_t)(b * DINNER + dd)) * HWSZ + tok_of(k, c0 + lc)];
        }
        __syncthreads();
#pragma unroll 4
        for (int lc = 0; lc < SCH; ++lc) {
            float dtraw = dtb;
#pragma unroll
            for (int r = 0; r < DTRANK; ++r) dtraw += s_bc[r * SCH + lc] * dtw[r];
            float delta = softplusf(dtraw);
            float u = s_u[d * SCH + lc];
            float du = delta * u;
            float y = 0.f;
#pragma unroll
            for (int n = 0; n < DSTATE; ++n) {
                float dA = __expf(delta * Av[n]);
                hst[n] = hst[n] * dA + du * s_bc[(DTRANK + n) * SCH + lc];
                y += hst[n] * s_bc[(DTRANK + DSTATE + n) * SCH + lc];
            }
            yo[c0 + lc] = y + Dd * u;
        }
        __syncthreads();
    }
}

// combine 4 directions -> token-major (t, d)
__global__ void combine_kernel(const float* __restrict__ ys,
                               float* __restrict__ out) {
    int i = blockIdx.x * blockDim.x + threadIdx.x;
    if (i >= NTOK * DINNER) return;
    int t = i / DINNER, d = i % DINNER;
    int b = t / HWSZ, hw = t % HWSZ;
    int h = hw / WWD, w = hw % WWD;
    int l1 = w * HH + h;
    const float* base = ys + (size_t)b * NDIR * DINNER * LSEQ;
    float y = base[((size_t)(0 * DINNER + d)) * LSEQ + hw]
            + base[((size_t)(2 * DINNER + d)) * LSEQ + (LSEQ - 1 - hw)]
            + base[((size_t)(1 * DINNER + d)) * LSEQ + l1]
            + base[((size_t)(3 * DINNER + d)) * LSEQ + (LSEQ - 1 - l1)];
    out[(size_t)t * DINNER + d] = y;
}

// layernorm(192) + silu(z) gating -> f16
__global__ void ln_gate_kernel(const float* __restrict__ y,
                               const float* __restrict__ g, const float* __restrict__ b,
                               const float* __restrict__ xz,
                               _Float16* __restrict__ out) {
    int t = blockIdx.x * blockDim.x + threadIdx.x;
    if (t >= NTOK) return;
    const float* row = y + (size_t)t * DINNER;
    float s = 0.f, s2 = 0.f;
    for (int c = 0; c < DINNER; ++c) { float x = row[c]; s += x; s2 += x * x; }
    float mu = s / DINNER;
    float inv = rsqrtf(s2 / DINNER - mu * mu + LN_EPS);
    const float* zrow = xz + (size_t)t * (2 * DINNER) + DINNER;
    for (int c = 0; c < DINNER; ++c) {
        float yn = ((row[c] - mu) * inv) * g[c] + b[c];
        out[(size_t)t * DINNER + c] = (_Float16)(yn * siluf(zrow[c]));
    }
}

// out_proj GEMM (96x192) -> f16
__global__ void gemm_out_kernel(const _Float16* __restrict__ A,
                                const _Float16* __restrict__ W,
                                _Float16* __restrict__ out) {
    const int lane = threadIdx.x & 31, wid = threadIdx.x >> 5;
    const int tilesN = DIMC / 16, tot = (NTOK / 16) * tilesN;
    int tile = blockIdx.x * 8 + wid;
    if (tile >= tot) return;
    int mt = tile / tilesN, nt = tile % tilesN;
    v8f acc = gemm_tile<DINNER>(A, W, mt, nt, lane);
    int col = nt * 16 + (lane & 15);
#pragma unroll
    for (int r = 0; r < 8; ++r) {
        int row = mt * 16 + r + ((lane >> 4) << 3);
        out[(size_t)row * DIMC + col] = (_Float16)acc[r];
    }
}

// post GEMM (96x96) + BN + GELU + gated residual -> d_out (B,C,H,W) f32
__global__ void gemm_post_kernel(const _Float16* __restrict__ A,
                                 const _Float16* __restrict__ W,
                                 const float* __restrict__ g, const float* __restrict__ b,
                                 const float* __restrict__ m, const float* __restrict__ v,
                                 const float* __restrict__ feature,
                                 const float* __restrict__ gate,
                                 float* __restrict__ out) {
    const int lane = threadIdx.x & 31, wid = threadIdx.x >> 5;
    const int tilesN = DIMC / 16, tot = (NTOK / 16) * tilesN;
    int tile = blockIdx.x * 8 + wid;
    if (tile >= tot) return;
    int mt = tile / tilesN, nt = tile % tilesN;
    v8f acc = gemm_tile<DIMC>(A, W, mt, nt, lane);
    int col = nt * 16 + (lane & 15);
    float sc = g[col] * rsqrtf(v[col] + BN_EPS);
    float gt = gate[0];
#pragma unroll
    for (int r = 0; r < 8; ++r) {
        int row = mt * 16 + r + ((lane >> 4) << 3);   // token index
        int bb = row / HWSZ, hw = row % HWSZ;
        float x = geluf((acc[r] - m[col]) * sc + b[col]);
        size_t idx = ((size_t)bb * DIMC + col) * HWSZ + hw;
        out[idx] = feature[idx] + gt * x;
    }
}

// ---------------------------------------------------------------- launch
extern "C" void kernel_launch(void* const* d_in, const int* in_sizes, int n_in,
                              void* d_out, int out_size, void* d_ws, size_t ws_size,
                              hipStream_t stream) {
    (void)in_sizes; (void)n_in; (void)out_size; (void)ws_size;
    const float* feature = (const float*)d_in[0];
    const float* pred    = (const float*)d_in[1];
    const float* bnp_g = (const float*)d_in[2];
    const float* bnp_b = (const float*)d_in[3];
    const float* bnp_m = (const float*)d_in[4];
    const float* bnp_v = (const float*)d_in[5];
    const float* pre_w = (const float*)d_in[6];
    const float* pre_g = (const float*)d_in[7];
    const float* pre_b = (const float*)d_in[8];
    const float* pre_m = (const float*)d_in[9];
    const float* pre_v = (const float*)d_in[10];
    const float* ln_g  = (const float*)d_in[11];
    const float* ln_b  = (const float*)d_in[12];
    const float* in_proj_w = (const float*)d_in[13];
    const float* conv_w = (const float*)d_in[14];
    const float* conv_b = (const float*)d_in[15];
    const float* x_proj_w = (const float*)d_in[16];
    const float* dt_w = (const float*)d_in[17];
    const float* dt_b = (const float*)d_in[18];
    const float* A_logs = (const float*)d_in[19];
    const float* Ds = (const float*)d_in[20];
    const float* on_g = (const float*)d_in[21];
    const float* on_b = (const float*)d_in[22];
    const float* out_proj_w = (const float*)d_in[23];
    const float* post_w = (const float*)d_in[24];
    const float* post_g = (const float*)d_in[25];
    const float* post_b = (const float*)d_in[26];
    const float* post_m = (const float*)d_in[27];
    const float* post_v = (const float*)d_in[28];
    const float* gate = (const float*)d_in[29];
    float* outp = (float*)d_out;

    // workspace carve-up (256B aligned)
    size_t off = 0;
    auto carve = [&](size_t bytes) -> char* {
        char* p = (char*)d_ws + off;
        off = (off + bytes + 255) & ~(size_t)255;
        return p;
    };
    _Float16* w16_pre  = (_Float16*)carve(DIMC * DIMC * 2);
    _Float16* w16_in   = (_Float16*)carve(2 * DINNER * DIMC * 2);
    _Float16* w16_out  = (_Float16*)carve(DIMC * DINNER * 2);
    _Float16* w16_post = (_Float16*)carve(DIMC * DIMC * 2);
    _Float16* fm16  = (_Float16*)carve((size_t)NTOK * DIMC * 2);
    float*    bufA  = (float*)carve((size_t)NTOK * DIMC * 4);
    _Float16* xcl16 = (_Float16*)carve((size_t)NTOK * DIMC * 2);
    float*    xz    = (float*)carve((size_t)NTOK * 2 * DINNER * 4);
    float*    xconv = (float*)carve((size_t)BATCH * DINNER * HWSZ * 4);
    float*    xdbl  = (float*)carve((size_t)BATCH * NDIR * CDBL * LSEQ * 4);
    float*    yscan = (float*)carve((size_t)BATCH * NDIR * DINNER * LSEQ * 4);
    float*    ycomb = (float*)carve((size_t)NTOK * DINNER * 4);
    _Float16* yg16  = (_Float16*)carve((size_t)NTOK * DINNER * 2);
    _Float16* yo16  = (_Float16*)carve((size_t)NTOK * DIMC * 2);

    const int TB = 256;
    auto cdiv = [](int a, int b) { return (a + b - 1) / b; };

    // weight conversions
    cvt_f16_kernel<<<cdiv(DIMC * DIMC, TB), TB, 0, stream>>>(pre_w, w16_pre, DIMC * DIMC);
    cvt_f16_kernel<<<cdiv(2 * DINNER * DIMC, TB), TB, 0, stream>>>(in_proj_w, w16_in, 2 * DINNER * DIMC);
    cvt_f16_kernel<<<cdiv(DIMC * DINNER, TB), TB, 0, stream>>>(out_proj_w, w16_out, DIMC * DINNER);
    cvt_f16_kernel<<<cdiv(DIMC * DIMC, TB), TB, 0, stream>>>(post_w, w16_post, DIMC * DIMC);

    // 1) reverse-attention mask
    mask_kernel<<<cdiv(NTOK * DIMC, TB), TB, 0, stream>>>(
        feature, pred, bnp_g, bnp_b, bnp_m, bnp_v, fm16);
    // 2) pre GEMM + BN + GELU
    gemm_pre_kernel<<<cdiv((NTOK / 16) * (DIMC / 16), 8), TB, 0, stream>>>(
        fm16, w16_pre, pre_g, pre_b, pre_m, pre_v, bufA);
    // 3) layernorm -> f16
    ln96_kernel<<<cdiv(NTOK, TB), TB, 0, stream>>>(bufA, ln_g, ln_b, xcl16);
    // 4) in_proj GEMM
    gemm_in_kernel<<<cdiv((NTOK / 16) * ((2 * DINNER) / 16), 8), TB, 0, stream>>>(
        xcl16, w16_in, xz);
    // 5) depthwise conv + SiLU
    conv_kernel<<<cdiv(BATCH * DINNER * HWSZ, TB), TB, 0, stream>>>(
        xz, conv_w, conv_b, xconv);
    // 6) x_proj -> dts/Bs/Cs
    xproj_kernel<<<cdiv(BATCH * NDIR * LSEQ, 128), 128, 0, stream>>>(
        xconv, x_proj_w, xdbl);
    // 7) selective scan (16 workgroups, one per (b,k))
    scan_kernel<<<BATCH * NDIR, DINNER, 0, stream>>>(
        xconv, xdbl, dt_w, dt_b, A_logs, Ds, yscan);
    // 8) combine 4 directions
    combine_kernel<<<cdiv(NTOK * DINNER, TB), TB, 0, stream>>>(yscan, ycomb);
    // 9) layernorm + SiLU(z) gate
    ln_gate_kernel<<<cdiv(NTOK, TB), TB, 0, stream>>>(ycomb, on_g, on_b, xz, yg16);
    // 10) out_proj GEMM
    gemm_out_kernel<<<cdiv((NTOK / 16) * (DIMC / 16), 8), TB, 0, stream>>>(
        yg16, w16_out, yo16);
    // 11) post GEMM + BN + GELU + gated residual
    gemm_post_kernel<<<cdiv((NTOK / 16) * (DIMC / 16), 8), TB, 0, stream>>>(
        yo16, w16_post, post_g, post_b, post_m, post_v, feature, gate, outp);
}
